// ProjectionBinarySearch_60653528154157
// MI455X (gfx1250) — matrix operand order repaired and checked by reading, not compiled, not run on hardware
//
#include <hip/hip_runtime.h>
#include <hip/hip_bf16.h>

// ---------------------------------------------------------------------------
// Problem constants (from reference)
// ---------------------------------------------------------------------------
#define BATCH 128
#define FDIM  12288
#define HDIM  2048
#define TAU_C 0.1f
#define THR_C 1e-3f
#define MAX_STEPS_C 30

#define KSPLIT   4
#define KCHUNK   (FDIM / KSPLIT)     // 3072
#define KSTEPS   (KCHUNK / 32)       // 96 K-steps of 32
#define NBLOCK   64                  // N columns per workgroup
#define NGRID    (HDIM / NBLOCK)     // 32 N-blocks

typedef __attribute__((ext_vector_type(4)))  __bf16 v4bf;
typedef __attribute__((ext_vector_type(8)))  __bf16 v8bf;
typedef __attribute__((ext_vector_type(16))) __bf16 v16bf;
typedef __attribute__((ext_vector_type(8)))  float  v8f;

// ---------------------------------------------------------------------------
// Kernel 1: UVp[ks] = [x0; x](256 x Kchunk) @ W1[Kchunk x 2048 slab]
// Workgroup = 8 waves computes 256x64 output block (W1 read ONCE from HBM).
// Wave tile = 64x32 (4 M-tiles x 2 N-tiles) -> 8 WMMAs per 32-wide K-step.
// LDS double-buffered bf16 staging; fragments read as 2x ds_load_b128 each.
// Grid: (32 N-blocks, 4 K-splits), 256 threads.
// ---------------------------------------------------------------------------
__global__ __launch_bounds__(256) void gemm_uv_kernel(
    const float* __restrict__ x0, const float* __restrict__ x,
    const float* __restrict__ W1, float* __restrict__ UVp)
{
    // A staged row-major [m][k] (bf16), B staged TRANSPOSED [n][k] (bf16):
    // makes every WMMA fragment two contiguous 16B chunks in LDS.
    __shared__ __bf16 As[2][256 * 32];   // 2 x 16 KB
    __shared__ __bf16 Bs[2][NBLOCK * 32];// 2 x  4 KB

    const int lane = threadIdx.x & 31;
    const int wave = threadIdx.x >> 5;
    const int lm   = lane & 15;          // row/col within 16x16 tile
    const int kh   = lane >> 4;          // K-half selector (this code's contiguous k = 16*kh + s)

    const int nB = blockIdx.x * NBLOCK;          // N-block origin
    const int kBase = blockIdx.y * KCHUNK;       // K-split origin
    const int mW = (wave & 3) * 64;              // wave M origin (4 M-tiles)
    const int nW = (wave >> 2) * 32;             // wave N origin (2 N-tiles)

    v8f acc[4][2] = {};

    // ---- stage one 32-wide K-slab into LDS buffer `b` -----------------
    auto stage = [&](int b, int k0) {
        // A slab: 256 rows x 32 cols, coalesced float4, cvt to bf16
#pragma unroll
        for (int i = 0; i < 8; ++i) {
            const int c   = threadIdx.x + i * 256;   // 2048 float4 chunks
            const int row = c >> 3;                  // 8 float4 per row
            const int c4  = c & 7;
            const float* src = (row < BATCH)
                ? (x0 + (size_t)row * FDIM)
                : (x  + (size_t)(row - BATCH) * FDIM);
            const float4 f = *(const float4*)(src + k0 + c4 * 4);
            v4bf o;
            o[0] = (__bf16)f.x; o[1] = (__bf16)f.y;
            o[2] = (__bf16)f.z; o[3] = (__bf16)f.w;
            *(v4bf*)&As[b][row * 32 + c4 * 4] = o;
        }
        // B slab: W1 rows k0..k0+31, cols nB..nB+63 -> transposed Bs[n][k]
#pragma unroll
        for (int i = 0; i < 2; ++i) {
            const int c  = threadIdx.x + i * 256;    // 512 float4 chunks
            const int k  = c >> 4;                   // 16 float4 per row
            const int c4 = c & 15;
            const float4 f = *(const float4*)(W1 + (size_t)(k0 + k) * HDIM + nB + c4 * 4);
            const int n0 = c4 * 4;
            Bs[b][(n0 + 0) * 32 + k] = (__bf16)f.x;
            Bs[b][(n0 + 1) * 32 + k] = (__bf16)f.y;
            Bs[b][(n0 + 2) * 32 + k] = (__bf16)f.z;
            Bs[b][(n0 + 3) * 32 + k] = (__bf16)f.w;
        }
    };

    // ---- consume buffer `b`: 12 ds_load_b128 -> 8 WMMAs ---------------
    auto compute = [&](int b) {
        v16bf bfrag[2];
#pragma unroll
        for (int ni = 0; ni < 2; ++ni) {
            const int n = nW + ni * 16 + lm;
            const v8bf lo = *(const v8bf*)&Bs[b][n * 32 + 16 * kh];
            const v8bf hi = *(const v8bf*)&Bs[b][n * 32 + 16 * kh + 8];
            bfrag[ni] = __builtin_shufflevector(lo, hi,
                0,1,2,3,4,5,6,7,8,9,10,11,12,13,14,15);
        }
#pragma unroll
        for (int mi = 0; mi < 4; ++mi) {
            const int m = mW + mi * 16 + lm;
            const v8bf lo = *(const v8bf*)&As[b][m * 32 + 16 * kh];
            const v8bf hi = *(const v8bf*)&As[b][m * 32 + 16 * kh + 8];
            const v16bf af = __builtin_shufflevector(lo, hi,
                0,1,2,3,4,5,6,7,8,9,10,11,12,13,14,15);
#pragma unroll
            for (int ni = 0; ni < 2; ++ni) {
                acc[mi][ni] = __builtin_amdgcn_wmma_f32_16x16x32_bf16(
                    false, af, false, bfrag[ni],
                    (short)0, acc[mi][ni], false, false);
            }
        }
    };

    stage(0, kBase);
    __syncthreads();

    for (int kk = 0; kk < KSTEPS; ++kk) {
        const int buf = kk & 1;
        if (kk + 1 < KSTEPS) {
            stage(buf ^ 1, kBase + (kk + 1) * 32);
        }
        if (kk + 2 < KSTEPS) {
            // gfx1250 global_prefetch_b8: warm L2 for the K+2 W1 slab
            const int kp = kBase + (kk + 2) * 32;
            __builtin_prefetch(W1 + (size_t)(kp + (threadIdx.x >> 3)) * HDIM
                               + nB + (threadIdx.x & 7) * 8, 0, 1);
        }
        compute(buf);
        __syncthreads();
    }

    // ---- store this K-split's partial: D-tile layout M = v + 8*kh -----
    float* __restrict__ P = UVp + (size_t)blockIdx.y * 256 * HDIM;
#pragma unroll
    for (int mi = 0; mi < 4; ++mi)
#pragma unroll
        for (int ni = 0; ni < 2; ++ni)
#pragma unroll
            for (int v = 0; v < 8; ++v) {
                const int m = mW + mi * 16 + 8 * kh + v;
                const int n = nB + nW + ni * 16 + lm;
                P[(size_t)m * HDIM + n] = acc[mi][ni][v];
            }
}

// ---------------------------------------------------------------------------
// Kernel 2: per-sample bisection on c.  f(c) = sum_h relu((1-c)u + c v) w2 - TAU
// Sums the 4 K-split partials while staging into LDS, then 31 reductions.
// ---------------------------------------------------------------------------
__global__ __launch_bounds__(256) void bisect_kernel(
    const float* __restrict__ UVp, const float* __restrict__ w2,
    float* __restrict__ cOut, int* __restrict__ doneOut)
{
    __shared__ float sU[HDIM];
    __shared__ float sV[HDIM];
    __shared__ float sW[HDIM];
    __shared__ float sred[8];

    const int s = blockIdx.x;
    const int t = threadIdx.x;

    for (int i = t; i < HDIM; i += 256) {
        float u = 0.0f, v = 0.0f;
#pragma unroll
        for (int j = 0; j < KSPLIT; ++j) {
            u += UVp[((size_t)j * 256 + s) * HDIM + i];
            v += UVp[((size_t)j * 256 + BATCH + s) * HDIM + i];
        }
        sU[i] = u;
        sV[i] = v;
        sW[i] = w2[i];
    }
    __syncthreads();

    auto feval = [&](float c) -> float {
        float part = 0.0f;
#pragma unroll
        for (int j = 0; j < 8; ++j) {
            const int i = t + j * 256;
            const float z = (1.0f - c) * sU[i] + c * sV[i];
            part += fmaxf(z, 0.0f) * sW[i];
        }
#pragma unroll
        for (int off = 16; off > 0; off >>= 1)
            part += __shfl_down(part, off, 32);
        if ((t & 31) == 0) sred[t >> 5] = part;
        __syncthreads();
        const float tot = sred[0] + sred[1] + sred[2] + sred[3] +
                          sred[4] + sred[5] + sred[6] + sred[7];
        __syncthreads();
        return tot - TAU_C;
    };

    const float f1 = feval(1.0f);
    const bool inside = (f1 <= 0.0f);

    float a = 0.0f, b = 1.0f, c = 1.0f;
    bool done = inside;

    for (int it = 0; it < MAX_STEPS_C; ++it) {
        const float cm = (a + b) * 0.5f;
        const float fv = feval(cm);          // uniform across block
        const bool upd  = !done;
        const bool hit  = upd && (fv >= -THR_C) && (fv < 0.0f);
        const bool go_a = upd && (fv < 0.0f) && !hit;
        const bool go_b = upd && (fv >= 0.0f);
        if (go_a) a = cm;
        if (go_b) b = cm;
        if (upd)  c = cm;
        done = done || hit;
    }
    if (inside) c = 1.0f;

    if (t == 0) {
        cOut[s]    = c;
        doneOut[s] = done ? 1 : 0;
    }
}

// ---------------------------------------------------------------------------
// Kernel 3: out[s,:] = done ? (1-c)*x0 + c*x : NaN
// ---------------------------------------------------------------------------
__global__ __launch_bounds__(256) void writeout_kernel(
    const float* __restrict__ x0, const float* __restrict__ x,
    const float* __restrict__ cArr, const int* __restrict__ doneArr,
    float* __restrict__ out)
{
    const size_t total = (size_t)BATCH * FDIM;
    const size_t idx = (size_t)blockIdx.x * blockDim.x + threadIdx.x;
    if (idx >= total) return;
    const int s = (int)(idx / FDIM);
    const float c = cArr[s];
    const bool  d = doneArr[s] != 0;
    const float p = (1.0f - c) * x0[idx] + c * x[idx];
    out[idx] = d ? p : __builtin_nanf("");
}

// ---------------------------------------------------------------------------
// Launcher. Inputs: d_in[0]=x0 (128x12288 f32), d_in[1]=x (128x12288 f32),
// d_in[2]=W1 (12288x2048 f32), d_in[3]=w2 (2048 f32). Output: 128x12288 f32.
// Workspace: UVp (4 x 256 x 2048 f32 = 8 MB) | c (128 f32) | done (128 i32).
// ---------------------------------------------------------------------------
extern "C" void kernel_launch(void* const* d_in, const int* in_sizes, int n_in,
                              void* d_out, int out_size, void* d_ws, size_t ws_size,
                              hipStream_t stream) {
    const float* x0 = (const float*)d_in[0];
    const float* x  = (const float*)d_in[1];
    const float* W1 = (const float*)d_in[2];
    const float* w2 = (const float*)d_in[3];
    float* out = (float*)d_out;

    float* UVp  = (float*)d_ws;                              // 4*256*2048 f32
    float* cArr = UVp + (size_t)KSPLIT * 256 * HDIM;         // 128 f32
    int*   dArr = (int*)(cArr + BATCH);                      // 128 i32

    // 1) GEMM with split-K: [x0;x] @ W1 -> 4 partial buffers
    dim3 grid(NGRID, KSPLIT);
    gemm_uv_kernel<<<grid, 256, 0, stream>>>(x0, x, W1, UVp);

    // 2) Per-sample bisection (sums partials while staging to LDS)
    bisect_kernel<<<BATCH, 256, 0, stream>>>(UVp, w2, cArr, dArr);

    // 3) Final projection + NaN fill
    const size_t total = (size_t)BATCH * FDIM;
    const int blocks = (int)((total + 255) / 256);
    writeout_kernel<<<blocks, 256, 0, stream>>>(x0, x, cArr, dArr, out);
}
